// Attention_66297115181568
// MI455X (gfx1250) — matrix operand order repaired and checked by reading, not compiled, not run on hardware
//
#include <hip/hip_runtime.h>
#include <hip/hip_bf16.h>
#include <math.h>

typedef __attribute__((ext_vector_type(16))) _Float16 v16h;
typedef __attribute__((ext_vector_type(8)))  _Float16 v8h;
typedef __attribute__((ext_vector_type(8)))  float    v8f;

#define Nn 512
#define DIMc 512
#define Hh 8
#define Dd 64
#define SSTR 520   // padded LDS stride for S rows

__device__ __forceinline__ v8f wmma_f16(v16h a, v16h b, v8f c) {
    return __builtin_amdgcn_wmma_f32_16x16x32_f16(false, a, false, b, (short)0, c, false, false);
}

__device__ __forceinline__ float wave_max(float v) {
#pragma unroll
    for (int o = 16; o > 0; o >>= 1) v = fmaxf(v, __shfl_xor(v, o, 32));
    return v;
}
__device__ __forceinline__ float wave_sum(float v) {
#pragma unroll
    for (int o = 16; o > 0; o >>= 1) v += __shfl_xor(v, o, 32);
    return v;
}

// A fragment (16x32 f16): src row-major f16, rows m0.., cols k0.. (two 16B loads)
__device__ __forceinline__ v16h frag_a_f16(const _Float16* __restrict__ src, int ld,
                                           int m0, int k0, int lane) {
    const int m  = lane & 15;
    const int kb = (lane < 16) ? 0 : 8;
    const v8h* p = reinterpret_cast<const v8h*>(src + (size_t)(m0 + m) * ld + k0 + kb);
    v8h lo = p[0];      // K = kb .. kb+7
    v8h hi = p[2];      // K = kb+16 .. kb+23   (+16 halves)
    v16h a;
#pragma unroll
    for (int t = 0; t < 8; ++t) { a[t] = lo[t]; a[t + 8] = hi[t]; }
    return a;
}

// B fragment (32x16 f16) from K-contiguous storage Bt[n][k] (two 16B loads)
__device__ __forceinline__ v16h frag_b_f16t(const _Float16* __restrict__ srcT, int ld,
                                            int k0, int n0, int lane) {
    const int n  = lane & 15;
    const int kb = (lane < 16) ? 0 : 8;
    const v8h* p = reinterpret_cast<const v8h*>(srcT + (size_t)(n0 + n) * ld + k0 + kb);
    v8h lo = p[0];
    v8h hi = p[2];
    v16h b;
#pragma unroll
    for (int t = 0; t < 8; ++t) { b[t] = lo[t]; b[t + 8] = hi[t]; }
    return b;
}

// ---------------------------------------------------------------------------
// Kernel 0: one-time f16 conversion / transpose of activations & weights
//   y==0: xh = f16(x)                 (row-major copy)
//   y==1..4: Wt[c][k] = f16(W[k][c])  (transpose: Wq, Wk, Wv, Wo)
// grid = (1024, 5), block = 256
// ---------------------------------------------------------------------------
__global__ __launch_bounds__(256)
void prep_f16_kernel(const float* __restrict__ x,
                     const float* __restrict__ Wq, const float* __restrict__ Wk,
                     const float* __restrict__ Wv, const float* __restrict__ Wo,
                     _Float16* __restrict__ xh,  _Float16* __restrict__ Wqt,
                     _Float16* __restrict__ Wkt, _Float16* __restrict__ Wvt,
                     _Float16* __restrict__ Wot) {
    const int which = blockIdx.y;
    const int idx = blockIdx.x * 256 + threadIdx.x;     // 0 .. 256K-1
    if (which == 0) {
        xh[idx] = (_Float16)x[idx];
        return;
    }
    const float* W  = (which == 1) ? Wq : (which == 2) ? Wk : (which == 3) ? Wv : Wo;
    _Float16*    Wt = (which == 1) ? Wqt : (which == 2) ? Wkt : (which == 3) ? Wvt : Wot;
    const int k = idx >> 9;       // row of W   (coalesced read along c)
    const int c = idx & 511;
    Wt[(size_t)c * DIMc + k] = (_Float16)W[(size_t)k * DIMc + c];
}

// ---------------------------------------------------------------------------
// Kernel 1: QKV projection (WMMA f16) + bias + RoPE
//   writes fp32 Q (for rel contraction), f16 Q [h][n][d], f16 K [h][n][d],
//   f16 V transposed [h][d][n]
// grid = (256, 3), block = 128 (4 waves, 1 wave per 16x16 output tile)
// ---------------------------------------------------------------------------
__global__ __launch_bounds__(128)
void qkv_rope_kernel(const _Float16* __restrict__ xh, const float* __restrict__ rope,
                     const _Float16* __restrict__ Wqt, const float* __restrict__ bq,
                     const _Float16* __restrict__ Wkt, const float* __restrict__ bk,
                     const _Float16* __restrict__ Wvt, const float* __restrict__ bv,
                     float* __restrict__ q32, _Float16* __restrict__ Qh,
                     _Float16* __restrict__ Kh, _Float16* __restrict__ Vt) {
    const int lane = threadIdx.x & 31;
    const int wid  = threadIdx.x >> 5;
    const int tile = blockIdx.x * 4 + wid;        // 0..1023
    const int m0   = (tile >> 5) << 4;            // seq tile
    const int c0   = (tile & 31) << 4;            // feature tile
    const int which = blockIdx.y;                 // 0=q 1=k 2=v

    const _Float16* Wt  = (which == 0) ? Wqt : (which == 1) ? Wkt : Wvt;
    const float*    bias = (which == 0) ? bq : (which == 1) ? bk : bv;

    v8f acc = {};
#pragma unroll 4
    for (int k0 = 0; k0 < DIMc; k0 += 32) {
        v16h a = frag_a_f16(xh, DIMc, m0, k0, lane);
        v16h b = frag_b_f16t(Wt, DIMc, k0, c0, lane);
        acc = wmma_f16(a, b, acc);
    }

    const int n  = lane & 15;
    const int c  = c0 + n;
    const int h  = c >> 6;
    const int d  = c & 63;
    const int mo = (lane < 16) ? 0 : 8;

#pragma unroll
    for (int r = 0; r < 8; ++r) {
        int m = m0 + r + mo;
        float v = acc[r] + bias[c];
        if (which < 2) {
            // RoPE: pair element d^1 sits in lane^1 (same column-parity layout)
            float partner = __shfl_xor(v, 1, 32);
            float fr = rope[m * Dd + d];
            float sv = sinf(fr), cv = cosf(fr);
            float rot = (d & 1) ? partner : -partner;   // even d: -t[d+1]; odd d: +t[d-1]
            v = v * cv + rot * sv;
        }
        if (which == 0) {
            size_t idx = ((size_t)h * Nn + m) * Dd + d;
            q32[idx] = v; Qh[idx] = (_Float16)v;
        } else if (which == 1) {
            Kh[((size_t)h * Nn + m) * Dd + d] = (_Float16)v;
        } else {
            Vt[((size_t)h * Dd + d) * Nn + m] = (_Float16)v;   // transposed store
        }
    }
}

// ---------------------------------------------------------------------------
// Kernel 2: per (head, 16-row i-tile) fused attention
//   S = scale*Q Kt (WMMA) + rel contraction (fp32 streaming) -> softmax -> P V (WMMA)
// grid = 256 (h*32 + itile), block = 256 (8 waves)
// ---------------------------------------------------------------------------
__global__ __launch_bounds__(256)
void attn_kernel(const float* __restrict__ q32, const _Float16* __restrict__ Qh,
                 const _Float16* __restrict__ Kh, const _Float16* __restrict__ Vt,
                 const float* __restrict__ rel, const unsigned char* __restrict__ mask,
                 _Float16* __restrict__ o16) {
    __shared__ float S[16 * SSTR];     // score rows (fp32)
    __shared__ float qt[16 * Dd];      // fp32 q tile for rel contraction
    __shared__ float comb[8 * 256];    // AV partial combine

    const int lane = threadIdx.x & 31;
    const int wid  = threadIdx.x >> 5;
    const int h    = blockIdx.x >> 5;
    const int i0   = (blockIdx.x & 31) << 4;

    // stage fp32 q tile
    for (int t = threadIdx.x; t < 16 * Dd; t += 256) {
        int i = t >> 6, d = t & 63;
        qt[t] = q32[((size_t)h * Nn + i0 + i) * Dd + d];
    }

    // ---- S = scale * q @ k^T : 32 j-tiles over 8 waves -----------------
    const float scale = 0.125f;  // D^-0.5
    const _Float16* QhH = Qh + (size_t)h * Nn * Dd;
    const _Float16* KhH = Kh + (size_t)h * Nn * Dd;
    for (int jt = wid; jt < 32; jt += 8) {
        const int j0 = jt << 4;
        v8f acc = {};
#pragma unroll
        for (int ks = 0; ks < 2; ++ks) {
            const int k0 = ks * 32;
            v16h a = frag_a_f16(QhH, Dd, i0, k0, lane);          // Q rows, d-contig
            v16h b = frag_b_f16t(KhH, Dd, k0, j0, lane);         // B[d][j]=K[j][d]
            acc = wmma_f16(a, b, acc);
        }
        int n = lane & 15, mo = (lane < 16) ? 0 : 8;
#pragma unroll
        for (int r = 0; r < 8; ++r) S[(r + mo) * SSTR + j0 + n] = acc[r] * scale;
    }
    __syncthreads();

    // ---- rel_pos contraction: stream 2 MB/block, coalesced float4 ------
    {
        const int dl = lane & 15, jj = lane >> 4;
        for (int i = 0; i < 16; ++i) {
            const float q0 = qt[i * Dd + dl * 4 + 0];
            const float q1 = qt[i * Dd + dl * 4 + 1];
            const float q2 = qt[i * Dd + dl * 4 + 2];
            const float q3 = qt[i * Dd + dl * 4 + 3];
            const float* base = rel + (((size_t)h * Nn + i0 + i) * Nn) * Dd;
#pragma unroll 4
            for (int it = 0; it < 32; ++it) {
                const int j = it * 16 + wid * 2 + jj;
                const float* addr = base + (size_t)j * Dd + dl * 4;
                __builtin_prefetch(addr + 128 * Dd, 0, 0);   // speculative HBM prefetch
                const float4 rv = *reinterpret_cast<const float4*>(addr);
                float p = fmaf(rv.x, q0, fmaf(rv.y, q1, fmaf(rv.z, q2, rv.w * q3)));
                p += __shfl_xor(p, 1, 32);
                p += __shfl_xor(p, 2, 32);
                p += __shfl_xor(p, 4, 32);
                p += __shfl_xor(p, 8, 32);
                if (dl == 0) S[i * SSTR + j] += p;
            }
        }
    }
    __syncthreads();

    // ---- masked softmax: 2 rows per wave -------------------------------
    const float NEG = -3.402823466e38f;
#pragma unroll
    for (int rr = 0; rr < 2; ++rr) {
        const int i = wid * 2 + rr;
        float mx = NEG;
        for (int j = lane; j < Nn; j += 32) {
            float v = mask[j] ? S[i * SSTR + j] : NEG;
            S[i * SSTR + j] = v;
            mx = fmaxf(mx, v);
        }
        mx = wave_max(mx);
        float sum = 0.f;
        for (int j = lane; j < Nn; j += 32) {
            float e = __expf(S[i * SSTR + j] - mx);
            S[i * SSTR + j] = e;
            sum += e;
        }
        sum = wave_sum(sum);
        float inv = 1.0f / sum;
        for (int j = lane; j < Nn; j += 32) S[i * SSTR + j] *= inv;
    }
    __syncthreads();

    // ---- O = P @ V : wave = (d-tile, j-half); combine halves via LDS ---
    {
        const int dt = wid & 3, halfsel = wid >> 2;
        const int d0 = dt * 16;
        const _Float16* VtH = Vt + (size_t)h * Dd * Nn;
        v8f acc = {};
#pragma unroll
        for (int ks = 0; ks < 8; ++ks) {
            const int jbase = halfsel * 256 + ks * 32;
            v16h a; {   // P from LDS fp32 -> f16 (unavoidable, small)
                int m = lane & 15, kb = (lane < 16) ? 0 : 8;
                const float* p = &S[m * SSTR + jbase + kb];
#pragma unroll
                for (int t = 0; t < 8; ++t) {
                    a[t]     = (_Float16)p[t];
                    a[t + 8] = (_Float16)p[16 + t];
                }
            }
            // B[j][d] = V[j][d] from transposed Vt[d][j]: j-contiguous
            v16h b = frag_b_f16t(VtH, Nn, jbase, d0, lane);
            acc = wmma_f16(a, b, acc);
        }
#pragma unroll
        for (int r = 0; r < 8; ++r) comb[wid * 256 + r * 32 + lane] = acc[r];
        __syncthreads();
        if (wid < 4) {
            int n = lane & 15, mo = (lane < 16) ? 0 : 8;
#pragma unroll
            for (int r = 0; r < 8; ++r) {
                float v = comb[wid * 256 + r * 32 + lane] + comb[(wid + 4) * 256 + r * 32 + lane];
                o16[(size_t)(i0 + r + mo) * (Hh * Dd) + h * Dd + wid * 16 + n] = (_Float16)v;
            }
        }
    }
}

// ---------------------------------------------------------------------------
// Kernel 3: out = O @ Wo + bo  (WMMA GEMM 512x512x512, all-f16 operands)
// grid = 256, block = 128
// ---------------------------------------------------------------------------
__global__ __launch_bounds__(128)
void out_proj_kernel(const _Float16* __restrict__ o16, const _Float16* __restrict__ Wot,
                     const float* __restrict__ bo, float* __restrict__ out) {
    const int lane = threadIdx.x & 31;
    const int wid  = threadIdx.x >> 5;
    const int tile = blockIdx.x * 4 + wid;
    const int m0   = (tile >> 5) << 4;
    const int c0   = (tile & 31) << 4;

    v8f acc = {};
#pragma unroll 4
    for (int k0 = 0; k0 < DIMc; k0 += 32) {
        v16h a = frag_a_f16(o16, DIMc, m0, k0, lane);
        v16h b = frag_b_f16t(Wot, DIMc, k0, c0, lane);
        acc = wmma_f16(a, b, acc);
    }
    int n = lane & 15, mo = (lane < 16) ? 0 : 8;
#pragma unroll
    for (int r = 0; r < 8; ++r)
        out[(size_t)(m0 + r + mo) * DIMc + c0 + n] = acc[r] + bo[c0 + n];
}

// ---------------------------------------------------------------------------
extern "C" void kernel_launch(void* const* d_in, const int* in_sizes, int n_in,
                              void* d_out, int out_size, void* d_ws, size_t ws_size,
                              hipStream_t stream) {
    const float*         x    = (const float*)d_in[0];
    const unsigned char* mask = (const unsigned char*)d_in[1];
    const float*         rope = (const float*)d_in[2];
    const float*         rel  = (const float*)d_in[3];
    const float*         Wq   = (const float*)d_in[4];
    const float*         bq   = (const float*)d_in[5];
    const float*         Wk   = (const float*)d_in[6];
    const float*         bk   = (const float*)d_in[7];
    const float*         Wv   = (const float*)d_in[8];
    const float*         bv   = (const float*)d_in[9];
    const float*         Wo   = (const float*)d_in[10];
    const float*         bo   = (const float*)d_in[11];
    float* out = (float*)d_out;

    char* ws = (char*)d_ws;
    // layout (bytes): q32 1MB | Qh 512K | Kh 512K | Vt 512K | o16 512K |
    //                 xh 512K | Wqt 512K | Wkt 512K | Wvt 512K | Wot 512K
    float*    q32 = (float*)(ws);
    _Float16* Qh  = (_Float16*)(ws + 0x100000);
    _Float16* Kh  = (_Float16*)(ws + 0x180000);
    _Float16* Vt  = (_Float16*)(ws + 0x200000);
    _Float16* o16 = (_Float16*)(ws + 0x280000);
    _Float16* xh  = (_Float16*)(ws + 0x300000);
    _Float16* Wqt = (_Float16*)(ws + 0x380000);
    _Float16* Wkt = (_Float16*)(ws + 0x400000);
    _Float16* Wvt = (_Float16*)(ws + 0x480000);
    _Float16* Wot = (_Float16*)(ws + 0x500000);

    dim3 gPrep(1024, 5, 1);
    prep_f16_kernel<<<gPrep, 256, 0, stream>>>(x, Wq, Wk, Wv, Wo,
                                               xh, Wqt, Wkt, Wvt, Wot);
    dim3 gA(256, 3, 1);
    qkv_rope_kernel<<<gA, 128, 0, stream>>>(xh, rope, Wqt, bq, Wkt, bk, Wvt, bv,
                                            q32, Qh, Kh, Vt);
    attn_kernel<<<256, 256, 0, stream>>>(q32, Qh, Kh, Vt, rel, mask, o16);
    out_proj_kernel<<<256, 128, 0, stream>>>(o16, Wot, bo, out);
}
